// Attention_63367947485679
// MI455X (gfx1250) — compile-verified, matrix operand order checked by reading, not run on hardware
//
#include <hip/hip_runtime.h>
#include <math.h>

// ---------------------------------------------------------------------------
// CDNA5 (gfx1250). All GEMMs via v_wmma_f32_16x16x32_bf16. K-loop is fully
// branch-free (K zero-padded to /32; M/N over-reads only touch rows/cols that
// are never stored) and LDS double-buffered: tile k+1 is staged while tile k
// feeds the WMMA pipe; one barrier per K-step.
// ---------------------------------------------------------------------------

typedef __bf16 bf16;
typedef __attribute__((ext_vector_type(16))) __bf16 v16bf;
typedef __attribute__((ext_vector_type(8)))  float  v8f;
typedef __attribute__((ext_vector_type(4)))  unsigned int v4u;
typedef unsigned long long u64t;

#define LDK   40        // padded LDS K-stride (bf16): 80B row = 20 banks
#define TILE  (64 * LDK)

// ---------------------------------------------------------------------------
// Batched WMMA GEMM: C[z] = A[z] @ B[z]
//   A: M x K (bf16; TRANSA -> A[m,k] = Aptr[k*lda+m])
//   B: K x N (bf16; TRANSB -> B[k,n] = Bptr[n*ldb+k])
//   C: M x N (OutT). K multiple of 32, operand K-extent zero-padded.
// Batch offset: off = (z/zdiv)*s0 + (z%zdiv)*s1 (elements).
// ---------------------------------------------------------------------------
template <bool TRANSA, bool TRANSB, typename OutT>
__global__ __launch_bounds__(256) void gemm_wmma(
    const bf16* __restrict__ A, const bf16* __restrict__ B, OutT* __restrict__ C,
    int M, int N, int K, int lda, int ldb, int ldc, int zdiv,
    long sA0, long sA1, long sB0, long sB1, long sC0, long sC1)
{
  __shared__ bf16 As[2 * TILE];   // ping-pong [m][k]
  __shared__ bf16 Bs[2 * TILE];   // ping-pong, transposed: [n][k]

  const int tid  = threadIdx.x;
  const int z    = blockIdx.z;
  A += (long)(z / zdiv) * sA0 + (long)(z % zdiv) * sA1;
  B += (long)(z / zdiv) * sB0 + (long)(z % zdiv) * sB1;
  C += (long)(z / zdiv) * sC0 + (long)(z % zdiv) * sC1;

  const int m_blk = blockIdx.y * 64;
  const int n_blk = blockIdx.x * 64;

  const int wave = tid >> 5;
  const int lane = tid & 31;
  const int m0   = (wave & 3) * 16;
  const int n0   = (wave >> 2) * 32;

  // ---- branch-free tile staging (16B chunk per thread per tile) ----
  auto stageA = [&](bf16* dst, int k0) {
    if (!TRANSA) {
      const int r  = tid >> 2;            // 0..63
      const int c0 = (tid & 3) << 3;      // 0,8,16,24
      const bf16* src = &A[(long)(m_blk + r) * lda + k0 + c0];
      union { v4u q; u64t d[2]; } t;
      t.d[0] = *(const u64t*)(src);
      t.d[1] = *(const u64t*)(src + 4);
      *(v4u*)&dst[r * LDK + c0] = t.q;
    } else {
      const int kk = tid & 31;            // 0..31
      const int mc = (tid >> 5) << 3;     // 0..56
      const bf16* src = &A[(long)(k0 + kk) * lda + m_blk + mc];
      union { u64t d[2]; bf16 h[8]; } t;
      t.d[0] = *(const u64t*)(src);
      t.d[1] = *(const u64t*)(src + 4);
#pragma unroll
      for (int j = 0; j < 8; ++j) dst[(mc + j) * LDK + kk] = t.h[j];
    }
  };
  auto stageB = [&](bf16* dst, int k0) {
    if (TRANSB) {
      const int n  = tid >> 2;
      const int c0 = (tid & 3) << 3;
      const bf16* src = &B[(long)(n_blk + n) * ldb + k0 + c0];
      union { v4u q; u64t d[2]; } t;
      t.d[0] = *(const u64t*)(src);
      t.d[1] = *(const u64t*)(src + 4);
      *(v4u*)&dst[n * LDK + c0] = t.q;
    } else {
      const int kk = tid & 31;
      const int nc = (tid >> 5) << 3;
      const bf16* src = &B[(long)(k0 + kk) * ldb + n_blk + nc];
      union { u64t d[2]; bf16 h[8]; } t;
      t.d[0] = *(const u64t*)(src);
      t.d[1] = *(const u64t*)(src + 4);
#pragma unroll
      for (int j = 0; j < 8; ++j) dst[(nc + j) * LDK + kk] = t.h[j];
    }
  };

  v8f acc0 = {};
  v8f acc1 = {};

  // prologue: stage first tile into buffer 0
  stageA(As, 0);
  stageB(Bs, 0);

  int cur = 0;
  for (int k0 = 0; k0 < K; k0 += 32) {
    __syncthreads();                       // buf[cur] visible to all waves
    const int nxt = cur ^ 1;
    if (k0 + 32 < K) {                     // overlap: stage k+1 while computing k
      stageA(&As[nxt * TILE], k0 + 32);
      stageB(&Bs[nxt * TILE], k0 + 32);
    }

    // ---- fragments off LDS buf[cur] (2 x ds_load_b128 each) ----
    // ISA 16-bit layout: lanes 0-15 -> row=lane, K {0..7,16..23};
    // lanes 16-31 -> row=lane-16, K {8..15,24..31}. B symmetric over N.
    const bf16* Ac = &As[cur * TILE];
    const bf16* Bc = &Bs[cur * TILE];
    const int rr = lane & 15;
    const int kb = (lane >> 4) << 3;       // 0 or 8
    union Frag { v16bf v; v4u q[2]; } fa, fb0, fb1;
    fa.q[0]  = *(const v4u*)&Ac[(m0 + rr) * LDK + kb];
    fa.q[1]  = *(const v4u*)&Ac[(m0 + rr) * LDK + kb + 16];
    fb0.q[0] = *(const v4u*)&Bc[(n0 + rr) * LDK + kb];
    fb0.q[1] = *(const v4u*)&Bc[(n0 + rr) * LDK + kb + 16];
    fb1.q[0] = *(const v4u*)&Bc[(n0 + 16 + rr) * LDK + kb];
    fb1.q[1] = *(const v4u*)&Bc[(n0 + 16 + rr) * LDK + kb + 16];

    acc0 = __builtin_amdgcn_wmma_f32_16x16x32_bf16(false, fa.v, false, fb0.v,
                                                   (short)0, acc0, false, false);
    acc1 = __builtin_amdgcn_wmma_f32_16x16x32_bf16(false, fa.v, false, fb1.v,
                                                   (short)0, acc1, false, false);
    cur = nxt;
  }

  // ---- store: VGPR r -> M = r (lanes 0-15) / r+8 (lanes 16-31), N = lane&15
#pragma unroll
  for (int r = 0; r < 8; ++r) {
    int m  = m_blk + m0 + r + ((lane >> 4) << 3);
    int n1 = n_blk + n0 + (lane & 15);
    int n2 = n1 + 16;
    if (m < M && n1 < N) C[(long)m * ldc + n1] = (OutT)acc0[r];
    if (m < M && n2 < N) C[(long)m * ldc + n2] = (OutT)acc1[r];
  }
}

// ---------------------------------------------------------------------------
__global__ __launch_bounds__(256) void cvt_f32_bf16(const float* __restrict__ in,
                                                    bf16* __restrict__ out, long n)
{
  long i = (long)blockIdx.x * 256 + threadIdx.x;
  long stride = (long)gridDim.x * 256;
  for (; i < n; i += stride) out[i] = (bf16)in[i];
}

// out[mat][r][c] = (r < rows_in) ? in[mat][r][c] : 0   (row-padded convert)
__global__ __launch_bounds__(256) void cvt_pad_rows(const float* __restrict__ in,
                                                    bf16* __restrict__ out,
                                                    int rows_in, int rows_pad,
                                                    long cols, long nmats)
{
  long total = nmats * rows_pad * cols;
  long i = (long)blockIdx.x * 256 + threadIdx.x;
  long stride = (long)gridDim.x * 256;
  for (; i < total; i += stride) {
    long c = i % cols;
    long r = (i / cols) % rows_pad;
    long m = i / (cols * rows_pad);
    out[i] = (r < rows_in) ? (bf16)in[(m * rows_in + r) * cols + c] : (bf16)0.0f;
  }
}

// KV_S[b][j*196+n][c] = That[b][n][j*256+c]   (That rows padded to 224)
__global__ __launch_bounds__(256) void reorder_kvs(const bf16* __restrict__ That,
                                                   bf16* __restrict__ kvs, long total)
{
  long i = (long)blockIdx.x * 256 + threadIdx.x;
  long stride = (long)gridDim.x * 256;
  for (; i < total; i += stride) {
    int c = (int)(i % 256);
    long t = (i / 256) % 784;
    long b = i / (256L * 784);
    int j = (int)(t / 196), n = (int)(t % 196);
    kvs[i] = That[(b * 224 + n) * 1024 + (long)j * 256 + c];
  }
}

// per-(b,h) inv_std = rsqrt(var + eps) over `count` contiguous elements
__global__ __launch_bounds__(256) void instnorm_stats(const float* __restrict__ X,
                                                      float* __restrict__ invstd,
                                                      long count, float eps)
{
  __shared__ float r1[256], r2[256];
  const float* x = X + (long)blockIdx.x * count;
  float s = 0.f, s2 = 0.f;
  for (long i = threadIdx.x; i < count; i += 256) {
    float v = x[i];
    s += v; s2 += v * v;
  }
  r1[threadIdx.x] = s; r2[threadIdx.x] = s2;
  __syncthreads();
  for (int off = 128; off > 0; off >>= 1) {
    if (threadIdx.x < off) {
      r1[threadIdx.x] += r1[threadIdx.x + off];
      r2[threadIdx.x] += r2[threadIdx.x + off];
    }
    __syncthreads();
  }
  if (threadIdx.x == 0) {
    float mu  = r1[0] / (float)count;
    float var = r2[0] / (float)count - mu * mu;
    invstd[blockIdx.x] = rsqrtf(var + eps);
  }
}

// Row softmax of (x * scale [* invstd[mat]]), f32 in, bf16 out.
// Output rows have Lp columns; cols [L, Lp) zero-filled (K padding).
// (instnorm's mean shift cancels inside softmax, so only inv_std is applied.)
__global__ __launch_bounds__(256) void softmax_rows(
    const float* __restrict__ X, bf16* __restrict__ Y,
    int L, int Lp, long xMat, long yMat, int rows_per_mat,
    float scale, const float* __restrict__ invstd)
{
  __shared__ float red[256];
  const long row = blockIdx.x;
  const long mat = row / rows_per_mat;
  const long r   = row % rows_per_mat;
  const float* x = X + mat * xMat + r * (long)L;
  bf16* y = Y + mat * yMat + r * (long)Lp;
  float s = scale;
  if (invstd) s *= invstd[mat];

  float mx = -INFINITY;
  for (int i = threadIdx.x; i < L; i += 256) mx = fmaxf(mx, x[i] * s);
  red[threadIdx.x] = mx; __syncthreads();
  for (int off = 128; off > 0; off >>= 1) {
    if (threadIdx.x < off)
      red[threadIdx.x] = fmaxf(red[threadIdx.x], red[threadIdx.x + off]);
    __syncthreads();
  }
  mx = red[0]; __syncthreads();

  float sum = 0.f;
  for (int i = threadIdx.x; i < L; i += 256) sum += __expf(x[i] * s - mx);
  red[threadIdx.x] = sum; __syncthreads();
  for (int off = 128; off > 0; off >>= 1) {
    if (threadIdx.x < off) red[threadIdx.x] += red[threadIdx.x + off];
    __syncthreads();
  }
  float inv = 1.0f / red[0];

  for (int i = threadIdx.x; i < L; i += 256)
    y[i] = (bf16)(__expf(x[i] * s - mx) * inv);
  for (int i = L + threadIdx.x; i < Lp; i += 256)
    y[i] = (bf16)0.0f;
}

// ---------------------------------------------------------------------------
extern "C" void kernel_launch(void* const* d_in, const int* in_sizes, int n_in,
                              void* d_out, int out_size, void* d_ws, size_t ws_size,
                              hipStream_t stream)
{
  (void)in_sizes; (void)n_in; (void)out_size; (void)ws_size;

  const int  Hh  = 4;
  const int  NTP = 224;                    // 196 padded to /32
  const long nEmbCp = 32L * NTP * 1024;    // 7,340,032
  const long nEmbSp = 32L * NTP * 256;     // 1,835,008
  const long nAttn  = 32L * 4 * 196 * 784; // 19,668,992

  const float* e_in[4] = {(const float*)d_in[0], (const float*)d_in[1],
                          (const float*)d_in[2], (const float*)d_in[3]};
  const float* embC_in = (const float*)d_in[4];
  const float* Wc_in[4] = {(const float*)d_in[5], (const float*)d_in[6],
                           (const float*)d_in[7], (const float*)d_in[8]};
  const float* Wq_in[4] = {(const float*)d_in[9], (const float*)d_in[10],
                           (const float*)d_in[11], (const float*)d_in[12]};
  const float* Wk_in = (const float*)d_in[13];
  const float* Wv_in = (const float*)d_in[14];
  const float* Wo_in[4] = {(const float*)d_in[15], (const float*)d_in[16],
                           (const float*)d_in[17], (const float*)d_in[18]};
  float* out = (float*)d_out;

  char* wsb = (char*)d_ws;
  size_t cur = 0;
  auto alloc = [&](long elems, int esz) -> void* {
    cur = (cur + 255) & ~(size_t)255;
    void* p = wsb + cur;
    cur += (size_t)elems * esz;
    return p;
  };
  bf16* embC = (bf16*)alloc(nEmbCp, 2);                       // [B,224,1024], pad rows 0
  bf16* e_bf[4]; for (int i = 0; i < 4; ++i) e_bf[i] = (bf16*)alloc(nEmbSp, 2);
  bf16* Wc_bf[4]; for (int i = 0; i < 4; ++i) Wc_bf[i] = (bf16*)alloc(1024L*1024, 2);
  bf16* qb = (bf16*)alloc(nEmbCp, 2);
  bf16* kb = (bf16*)alloc(nEmbCp, 2);
  bf16* vb = (bf16*)alloc(nEmbCp, 2);
  float* attnF = (float*)alloc(nAttn, 4);       // also satat scores [128,196,196]
  bf16* probs  = (bf16*)alloc(nAttn, 2);        // satat: [128,224,224]; branch: [128,196,784]
  bf16* ctxb   = (bf16*)alloc(nEmbCp, 2);       // [B,224,1024]
  bf16* Thatb  = (bf16*)alloc(nEmbCp, 2);       // [B,224,1024]
  bf16* kvs    = (bf16*)alloc(32L*784*256, 2);  // [B,784,256]
  bf16* Wkb    = (bf16*)alloc(784L*784, 2);
  bf16* Wvb    = (bf16*)alloc(784L*784, 2);
  bf16* Kmix   = (bf16*)alloc(32L*784*256, 2);
  bf16* Vmix   = (bf16*)alloc(32L*784*256, 2);
  bf16* Wq_bf[4]; for (int i = 0; i < 4; ++i) Wq_bf[i] = (bf16*)alloc((long)NTP*196, 2);
  bf16* Wo_bf[4]; for (int i = 0; i < 4; ++i) Wo_bf[i] = (bf16*)alloc(256L*256, 2);
  bf16* Qmix   = (bf16*)alloc(nEmbSp, 2);       // [B,224,256]
  bf16* ctx2   = (bf16*)alloc(32L*196*256, 2);  // [B,196,256] compact
  float* invstd = (float*)alloc(128, 4);
  (void)alloc(1L << 20, 2);                     // 2MB guard for M/N over-reads

  dim3 blk(256);
  const int CVT_GRID = 2048;

  // ---- converts (zero row-padding where the dim later acts as K) ----
  cvt_pad_rows<<<CVT_GRID, blk, 0, stream>>>(embC_in, embC, 196, NTP, 1024, 32);
  for (int i = 0; i < 4; ++i)
    cvt_pad_rows<<<CVT_GRID, blk, 0, stream>>>(e_in[i], e_bf[i], 196, NTP, 256, 32);
  for (int i = 0; i < 4; ++i)
    cvt_pad_rows<<<CVT_GRID, blk, 0, stream>>>(Wq_in[i], Wq_bf[i], 196, NTP, 196, 1);
  for (int i = 0; i < 4; ++i)
    cvt_f32_bf16<<<CVT_GRID, blk, 0, stream>>>(Wc_in[i], Wc_bf[i], 1024L*1024);
  cvt_f32_bf16<<<CVT_GRID, blk, 0, stream>>>(Wk_in, Wkb, 784L*784);
  cvt_f32_bf16<<<CVT_GRID, blk, 0, stream>>>(Wv_in, Wvb, 784L*784);
  for (int i = 0; i < 4; ++i)
    cvt_f32_bf16<<<CVT_GRID, blk, 0, stream>>>(Wo_in[i], Wo_bf[i], 256L*256);

  // ================= SelfAttentionTAT over emb_C =================
  // q/k/v = embC @ W : [7168,1024]x[1024,1024]; zero pad rows propagate.
  {
    dim3 g(16, 112, 1);
    bf16* dst[3] = {qb, kb, vb};
    for (int i = 0; i < 3; ++i)
      gemm_wmma<false, false, bf16><<<g, blk, 0, stream>>>(
          embC, Wc_bf[i], dst[i], 7168, 1024, 1024, 1024, 1024, 1024,
          1, 0, 0, 0, 0, 0, 0);
  }
  // scores[b,h] = Qh @ Kh^T : M=196 N=196 K=256, z = b*4+h
  {
    dim3 g(4, 4, 128);
    gemm_wmma<false, true, float><<<g, blk, 0, stream>>>(
        qb, kb, attnF, 196, 196, 256, 1024, 1024, 196, Hh,
        (long)NTP*1024, 256, (long)NTP*1024, 256, 4L*196*196, 196L*196);
  }
  // softmax(1/16), write [224x224] rows with zeroed K-pad columns
  softmax_rows<<<128 * 196, blk, 0, stream>>>(attnF, probs, 196, NTP,
                                              196L*196, (long)NTP*NTP, 196,
                                              0.0625f, nullptr);
  // ctx[b,h] = P @ Vh : K=224 (P cols zero-padded, vb pad rows zero)
  {
    dim3 g(4, 4, 128);
    gemm_wmma<false, false, bf16><<<g, blk, 0, stream>>>(
        probs, vb, ctxb, 196, 256, NTP, NTP, 1024, 1024, Hh,
        4L*NTP*NTP, (long)NTP*NTP, (long)NTP*1024, 256, (long)NTP*1024, 256);
  }
  // T_hat = ctx @ Wo_c : [7168,1024]x[1024,1024]
  {
    dim3 g(16, 112, 1);
    gemm_wmma<false, false, bf16><<<g, blk, 0, stream>>>(
        ctxb, Wc_bf[3], Thatb, 7168, 1024, 1024, 1024, 1024, 1024,
        1, 0, 0, 0, 0, 0, 0);
  }
  // KV_S reshuffle [B,784,256]
  reorder_kvs<<<CVT_GRID, blk, 0, stream>>>(Thatb, kvs, 32L*784*256);

  // K/V token mix: A = W^T (trans), per batch b
  {
    dim3 g(4, 13, 32);
    gemm_wmma<true, false, bf16><<<g, blk, 0, stream>>>(
        Wkb, kvs, Kmix, 784, 256, 784, 784, 256, 256, 1,
        0, 0, 784L*256, 0, 784L*256, 0);
    gemm_wmma<true, false, bf16><<<g, blk, 0, stream>>>(
        Wvb, kvs, Vmix, 784, 256, 784, 784, 256, 256, 1,
        0, 0, 784L*256, 0, 784L*256, 0);
  }

  // ================= 4 branches =================
  for (int br = 0; br < 4; ++br) {
    // Q token mix: K=224 over zero-padded Wq rows / emb rows
    {
      dim3 g(4, 4, 32);
      gemm_wmma<true, false, bf16><<<g, blk, 0, stream>>>(
          Wq_bf[br], e_bf[br], Qmix, 196, 256, NTP, 196, 256, 256, 1,
          0, 0, (long)NTP*256, 0, (long)NTP*256, 0);
    }
    // attn[b,h] = Qh @ Kh^T : M=196 N=784 K=64
    {
      dim3 g(13, 4, 128);
      gemm_wmma<false, true, float><<<g, blk, 0, stream>>>(
          Qmix, Kmix, attnF, 196, 784, 64, 256, 256, 784, Hh,
          (long)NTP*256, 64, 784L*256, 64, 4L*196*784, 196L*784);
    }
    // instance-norm stats (mean shift cancels in softmax)
    instnorm_stats<<<128, blk, 0, stream>>>(attnF, invstd, 196L*784, 1e-5f);
    softmax_rows<<<128 * 196, blk, 0, stream>>>(attnF, probs, 784, 784,
                                                196L*784, 196L*784, 196,
                                                1.0f, invstd);
    // ctx[b,h] = P @ Vh : M=196 N=64 K=784 -> compact [B,196,256]
    {
      dim3 g(1, 4, 128);
      gemm_wmma<false, false, bf16><<<g, blk, 0, stream>>>(
          probs, Vmix, ctx2, 196, 64, 784, 784, 256, 256, Hh,
          4L*196*784, 196L*784, 784L*256, 64, 196L*256, 64);
    }
    // O = ctx @ Wo : [6272,256]x[256,256] -> f32 output
    {
      dim3 g(4, 98, 1);
      gemm_wmma<false, false, float><<<g, blk, 0, stream>>>(
          ctx2, Wo_bf[br], out + (long)br * 6272 * 256,
          6272, 256, 256, 256, 256, 256, 1, 0, 0, 0, 0, 0, 0);
    }
  }
}
